// VoxelPatchSuperResNetwork_54314156425266
// MI455X (gfx1250) — matrix operand order repaired and checked by reading, not compiled, not run on hardware
//
#include <hip/hip_runtime.h>
#include <cstddef>
#include <cstdint>

// ---------------------------------------------------------------------------
// CDNA5 / gfx1250 VoxelPatchSuperResNetwork forward.
// Every conv layer (incl. transposed convs via sub-pixel parity decomposition)
// runs through ONE implicit-GEMM kernel on v_wmma_f32_16x16x32_f16:
//   A (16x32 f16)   = weight tile staged in LDS (shared by 8 waves)
//   B (32x16 f16)x2 = im2col gather via precomputed LDS offset table,
//                     halo-padded activations -> zero branches, zero divides
//   C/D (16x16 f32) = accumulators; BN + LeakyReLU / sigmoid fused in epilogue
// Intermediates stored f16 (23.3 TB/s HBM roofline: halves traffic and feeds
// the f16 WMMA pipe directly).
// ---------------------------------------------------------------------------

typedef __attribute__((ext_vector_type(16))) _Float16 v16h;
typedef __attribute__((ext_vector_type(8)))  _Float16 v8h;
typedef __attribute__((ext_vector_type(8)))  float    v8f;

struct ConvParams {
    const _Float16* in;      // halo-padded input activations [n][Cin][Dp^3]
    const _Float16* w;       // repacked weights [Mtiles*16][Kpad], zero padded
    const int*      tab;     // K-row -> linear offset table [Kpad]
    const float*    scale;   // per out-channel scale (BN folded)
    const float*    shift;   // per out-channel shift (BN + conv bias folded)
    _Float16*       outh;    // optional f16 padded output
    float*          outf;    // optional f32 dense output (pre-activation)
    float*          outf_sig;// optional f32 dense sigmoid output
    _Float16*       outh_sig;// optional f16 padded sigmoid output
    int n_batch;
    int Cin, Cout, Mtiles;
    int Ws, Hs, Ds;          // output-subgrid dims handled by this launch
    int Dp;                  // input padded cube side (Din + 2)
    int Kpad;                // K rounded up to multiple of 32
    int padz, pady, padx;    // per-dim input padding
    int sm, oz, oy, ox;      // output mapping: out = sub*sm + off (parity)
    int Do_, Dpo;            // output cube side (dense) and padded side
    int lrelu;
};

template <int NX>
__global__ __launch_bounds__(256)
void conv3d_wmma_kernel(ConvParams p) {
    extern __shared__ char smem_raw[];
    _Float16* smw  = (_Float16*)smem_raw;                 // 16 x Kpad halves
    int*      stab = (int*)(smem_raw + 32 * p.Kpad);      // Kpad ints

    const int tid  = threadIdx.x;
    const int wave = tid >> 5;
    const int lane = tid & 31;

    const int xs    = p.Ws >> (NX == 2 ? 5 : 4);  // x super-tiles per row
    const int ysper = 8 / xs;                     // y rows per block

    int b = blockIdx.x;
    const int mt = b % p.Mtiles;  b /= p.Mtiles;
    const int yb = b % (p.Hs / ysper); b /= (p.Hs / ysper);
    const int z  = b % p.Ds;
    const int n  = b / p.Ds;

    // ---- cooperative staging: weight tile + offset table into LDS ----
    const int totalh = 16 * p.Kpad;
    const _Float16* wsrc = p.w + (size_t)(mt * 16) * p.Kpad;
    for (int i = tid * 8; i < totalh; i += 256 * 8)
        *(v8h*)(smw + i) = *(const v8h*)(wsrc + i);
    for (int i = tid; i < p.Kpad; i += 256)
        stab[i] = p.tab[i];
    __syncthreads();

    const int xt  = wave % xs;
    const int yi  = wave / xs;
    const int y   = yb * ysper + yi;
    const int x0  = xt * (16 * NX);
    const int col = lane & 15;
    const int hl  = lane >> 4;

    // per-lane base into the halo-padded input (no bounds checks needed)
    const int base = ((n * p.Cin * p.Dp + (z - p.padz + 1)) * p.Dp +
                      (y - p.pady + 1)) * p.Dp + (x0 + col - p.padx + 1);

    __builtin_prefetch(p.in + base, 0, 3);

    v8f acc[NX];
    #pragma unroll
    for (int u = 0; u < NX; ++u)
        acc[u] = (v8f){0.f, 0.f, 0.f, 0.f, 0.f, 0.f, 0.f, 0.f};

    const int nchunks = p.Kpad >> 5;
    for (int c = 0; c < nchunks; ++c) {
        const int kb = c << 5;

        // A fragment (16x32 f16) from LDS: lane holds row (lane&15);
        // hl=0 -> K 0..7 & 16..23 ; hl=1 -> K 8..15 & 24..31
        union { v16h v; v8h h[2]; } A;
        const _Float16* row = smw + col * p.Kpad + kb + hl * 8;
        A.h[0] = *(const v8h*)(row);
        A.h[1] = *(const v8h*)(row + 16);

        // offset-table entries for this lane's K rows: hl*8+0..7 & 16+hl*8+0..7 (b128 x4)
        union { int4 q[4]; int e[16]; } L;
        const int4* tp = (const int4*)(stab + kb + hl * 8);
        L.q[0] = tp[0]; L.q[1] = tp[1]; L.q[2] = tp[4]; L.q[3] = tp[5];

        // B fragments: column = x position (contiguous across lanes)
        union { v16h v; _Float16 e[16]; } B[NX];
        #pragma unroll
        for (int j = 0; j < 16; ++j) {
            const int lin = L.e[j];
            B[0].e[j] = p.in[base + lin];
            if (NX == 2) B[1].e[j] = p.in[base + 16 + lin];
        }

        acc[0] = __builtin_amdgcn_wmma_f32_16x16x32_f16(
                     false, A.v, false, B[0].v, (short)0, acc[0], false, false);
        if (NX == 2)
            acc[1] = __builtin_amdgcn_wmma_f32_16x16x32_f16(
                         false, A.v, false, B[1].v, (short)0, acc[1], false, false);
    }

    // ---- fused epilogue: BN scale/shift, LeakyReLU, sigmoid, stores ----
    const int zo = z * p.sm + p.oz;
    const int yo = y * p.sm + p.oy;
    #pragma unroll
    for (int u = 0; u < NX; ++u) {
        const int xo = (x0 + u * 16 + col) * p.sm + p.ox;
        #pragma unroll
        for (int i = 0; i < 8; ++i) {
            const int oc = mt * 16 + hl * 8 + i;
            if (oc < p.Cout) {
                float v = acc[u][i] * p.scale[oc] + p.shift[oc];
                if (p.lrelu) v = (v >= 0.f) ? v : 0.2f * v;
                if (p.outh || p.outh_sig) {
                    const size_t pidx =
                        ((size_t)(n * p.Cout + oc) * p.Dpo + (zo + 1)) * p.Dpo * p.Dpo +
                        (size_t)(yo + 1) * p.Dpo + (xo + 1);
                    if (p.outh) p.outh[pidx] = (_Float16)v;
                    if (p.outh_sig)
                        p.outh_sig[pidx] = (_Float16)(1.f / (1.f + expf(-v)));
                }
                if (p.outf || p.outf_sig) {
                    const size_t didx =
                        (((size_t)(n * p.Cout + oc) * p.Do_ + zo) * p.Do_ + yo) * p.Do_ + xo;
                    if (p.outf) p.outf[didx] = v;
                    if (p.outf_sig) p.outf_sig[didx] = 1.f / (1.f + expf(-v));
                }
            }
        }
    }
}

// ---------------------------------------------------------------------------
// Prep kernels
// ---------------------------------------------------------------------------

// K-row -> padded linear offset table (divisions live here, off the hot path)
__global__ void prep_tab(int* dst, int K, int Kpad, int k, int Dp) {
    const int i = blockIdx.x * blockDim.x + threadIdx.x;
    if (i >= Kpad) return;
    int lin = 0;
    if (i < K) {
        const int k3 = k * k * k;
        const int ic = i / k3, t = i - ic * k3;
        const int dz = t / (k * k), dy = (t / k) % k, dx = t % k;
        lin = ((ic * Dp + dz) * Dp + dy) * Dp + dx;
    }
    dst[i] = lin;
}

// Normal conv (src [Cout][Cin][k^3]) or stride-1 convT (src [Cin][Cout][k^3],
// kernel flipped) -> f16 rows [rows][Kpad], zero padded.
__global__ void prep_wconv(const float* src, _Float16* dst, int Cout, int Cin,
                           int k, int Kpad, int rows, int transposed) {
    const int i = blockIdx.x * blockDim.x + threadIdx.x;
    if (i >= rows * Kpad) return;
    const int k3 = k * k * k;
    const int r = i / Kpad, kk = i - r * Kpad;
    float v = 0.f;
    if (r < Cout && kk < Cin * k3) {
        const int ic = kk / k3, t = kk - ic * k3;
        if (transposed) {
            const int dz = t / (k * k), dy = (t / k) % k, dx = t % k;
            const int ft = ((k - 1 - dz) * k + (k - 1 - dy)) * k + (k - 1 - dx);
            v = src[(ic * Cout + r) * k3 + ft];
        } else {
            v = src[(r * Cin + ic) * k3 + t];
        }
    }
    dst[i] = (_Float16)v;
}

// Stride-2 convT (k=4,p=1) parity decomposition: 8 sub-kernels of 2x2x2 taps.
// src: [Cin][Cout][4][4][4]; dst: [8 parities][rows][Kpad].
__global__ void prep_wconvt2(const float* src, _Float16* dst, int Cout, int Cin,
                             int Kpad, int rows) {
    const int i = blockIdx.x * blockDim.x + threadIdx.x;
    const int per = rows * Kpad;
    if (i >= 8 * per) return;
    const int par = i / per, rem = i - par * per;
    const int r = rem / Kpad, kk = rem - r * Kpad;
    const int pz = (par >> 2) & 1, py = (par >> 1) & 1, px = par & 1;
    float v = 0.f;
    if (r < Cout && kk < Cin * 8) {
        const int ic = kk >> 3, t = kk & 7;
        const int tz = (t >> 2) & 1, ty = (t >> 1) & 1, tx = t & 1;
        const int kz = (3 - pz) - 2 * tz;
        const int ky = (3 - py) - 2 * ty;
        const int kx = (3 - px) - 2 * tx;
        v = src[(ic * Cout + r) * 64 + (kz * 16 + ky * 4 + kx)];
    }
    dst[i] = (_Float16)v;
}

__global__ void prep_bn(const float* g, const float* beta, const float* mean,
                        const float* var, const float* bconv,
                        float* scale, float* shift, int C, int Cpad) {
    const int i = blockIdx.x * blockDim.x + threadIdx.x;
    if (i >= Cpad) return;
    if (i < C) {
        const float s = g[i] * rsqrtf(var[i] + 1e-5f);
        scale[i] = s;
        shift[i] = (bconv[i] - mean[i]) * s + beta[i];
    } else { scale[i] = 0.f; shift[i] = 0.f; }
}

__global__ void prep_bias(const float* bconv, float* scale, float* shift,
                          int C, int Cpad) {
    const int i = blockIdx.x * blockDim.x + threadIdx.x;
    if (i >= Cpad) return;
    if (i < C) { scale[i] = 1.f; shift[i] = bconv[i]; }
    else       { scale[i] = 0.f; shift[i] = 0.f; }
}

// f32 dense 16^3 -> f16 padded 18^3
__global__ void cvt_in_pad(const float* src, _Float16* dst, int NB) {
    const int i = blockIdx.x * blockDim.x + threadIdx.x;
    if (i >= NB * 4096) return;
    const int n = i >> 12, r = i & 4095;
    const int z = r >> 8, y = (r >> 4) & 15, x = r & 15;
    dst[(size_t)n * 5832 + (z + 1) * 324 + (y + 1) * 18 + (x + 1)] = (_Float16)src[i];
}

__global__ void fill_f16v(_Float16* p, int n8) {   // zero n8*8 halves (b128)
    const int i = blockIdx.x * blockDim.x + threadIdx.x;
    if (i < n8) {
        v8h z = {(_Float16)0, (_Float16)0, (_Float16)0, (_Float16)0,
                 (_Float16)0, (_Float16)0, (_Float16)0, (_Float16)0};
        ((v8h*)p)[i] = z;
    }
}

__global__ void fill_i32(int* p, int v, int n) {
    const int i = blockIdx.x * blockDim.x + threadIdx.x;
    if (i < n) p[i] = v;
}

// ---------------------------------------------------------------------------
// Deterministic point-cloud scatter (max point index wins per cell)
// ---------------------------------------------------------------------------
__device__ __forceinline__ bool point_cell(const float* pc, const int* patch_idx,
                                           int b, int i, int* cell_out) {
    const float p0 = pc[i * 3 + 0] + 0.5f;
    const float p1 = pc[i * 3 + 1] + 0.5f;
    const float p2 = pc[i * 3 + 2] + 0.5f;
    const int pi = patch_idx[b];
    const int i0 = ((pi >> 4) & 3) * 64;
    const int i1 = ((pi >> 2) & 3) * 64;
    const int i2 = (pi & 3) * 64;
    const int q0 = (int)(p0 * 256.f);
    const int q1 = (int)(p1 * 256.f);
    const int q2 = (int)(p2 * 256.f);
    if (!(q0 >= i0 && q0 < i0 + 64 && q1 >= i1 && q1 < i1 + 64 &&
          q2 >= i2 && q2 < i2 + 64)) return false;
    const int c0 = (int)((p0 - (float)i0 * (1.f / 256.f)) * 255.996f);
    const int c1 = (int)((p1 - (float)i1 * (1.f / 256.f)) * 255.996f);
    const int c2 = (int)((p2 - (float)i2 * (1.f / 256.f)) * 255.996f);
    *cell_out = (c0 * 64 + c1) * 64 + c2;
    return true;
}

__global__ void scatter_pass1(const float* pc, const int* patch_idx,
                              int* winner, int Np, int NB) {
    const int gid = blockIdx.x * blockDim.x + threadIdx.x;
    if (gid >= Np * NB) return;
    const int b = gid / Np, i = gid - b * Np;
    int cell;
    if (point_cell(pc, patch_idx, b, i, &cell))
        atomicMax(&winner[b * 262144 + cell], i);
}

__global__ void scatter_pass2(const float* pc, const int* patch_idx,
                              const int* winner, const float* sig_f32,
                              _Float16* upd_h, int Np, int NB) {
    const int gid = blockIdx.x * blockDim.x + threadIdx.x;
    if (gid >= Np * NB) return;
    const int b = gid / Np, i = gid - b * Np;
    int cell;
    if (point_cell(pc, patch_idx, b, i, &cell)) {
        if (winner[b * 262144 + cell] == i) {
            const float occ = pc[i] + 0.5f;         // occ = flat(pc+0.5)[:N]
            const float old = sig_f32[b * 262144 + cell];
            const int c0 = cell >> 12, c1 = (cell >> 6) & 63, c2 = cell & 63;
            upd_h[(size_t)b * 287496 + (c0 + 1) * 4356 + (c1 + 1) * 66 + (c2 + 1)] =
                (_Float16)(old * 0.3f + occ * 0.7f);
        }
    }
}

// ---------------------------------------------------------------------------
// Host orchestration
// ---------------------------------------------------------------------------
static inline void launch_conv(const ConvParams& cp, hipStream_t stream) {
    const size_t sh = (size_t)36 * cp.Kpad;          // 32*Kpad wgt + 4*Kpad tab
    if (cp.Ws >= 32) {
        const int xs = cp.Ws >> 5, ysper = 8 / xs;
        const int grid = cp.Mtiles * (cp.Hs / ysper) * cp.Ds * cp.n_batch;
        conv3d_wmma_kernel<2><<<grid, 256, sh, stream>>>(cp);
    } else {
        const int xs = cp.Ws >> 4, ysper = 8 / xs;
        const int grid = cp.Mtiles * (cp.Hs / ysper) * cp.Ds * cp.n_batch;
        conv3d_wmma_kernel<1><<<grid, 256, sh, stream>>>(cp);
    }
}

static inline void fill_h(_Float16* p, size_t nelem, hipStream_t stream) {
    const int n8 = (int)(nelem / 8);
    fill_f16v<<<(n8 + 255) / 256, 256, 0, stream>>>(p, n8);
}

extern "C" void kernel_launch(void* const* d_in, const int* in_sizes, int n_in,
                              void* d_out, int out_size, void* d_ws, size_t ws_size,
                              hipStream_t stream) {
    (void)n_in; (void)out_size; (void)ws_size;
    const float* vox_patch = (const float*)d_in[0];
    const float* pc        = (const float*)d_in[1];
    // d_in[2] = pc_occ : unused by the reference forward (occ derives from pc)
    const int*   patch_idx = (const int*)d_in[3];
    // params flattened in JAX pytree (sorted-key) order; bn tuples = (g,b,m,v)
    const float* g1_b = (const float*)d_in[4];
    const float* g1_g = (const float*)d_in[5],  *g1_be = (const float*)d_in[6];
    const float* g1_m = (const float*)d_in[7],  *g1_v  = (const float*)d_in[8];
    const float* g1_w = (const float*)d_in[9];
    const float* g2_b = (const float*)d_in[10];
    const float* g2_g = (const float*)d_in[11], *g2_be = (const float*)d_in[12];
    const float* g2_m = (const float*)d_in[13], *g2_v  = (const float*)d_in[14];
    const float* g2_w = (const float*)d_in[15];
    const float* g3_b = (const float*)d_in[16];
    const float* g3_g = (const float*)d_in[17], *g3_be = (const float*)d_in[18];
    const float* g3_m = (const float*)d_in[19], *g3_v  = (const float*)d_in[20];
    const float* g3_w = (const float*)d_in[21];
    const float* g4_b = (const float*)d_in[22];
    const float* g4_w = (const float*)d_in[23];
    const float* r1_b = (const float*)d_in[24];
    const float* r1_g = (const float*)d_in[25], *r1_be = (const float*)d_in[26];
    const float* r1_m = (const float*)d_in[27], *r1_v  = (const float*)d_in[28];
    const float* r1_w = (const float*)d_in[29];
    const float* r2_b = (const float*)d_in[30];
    const float* r2_g = (const float*)d_in[31], *r2_be = (const float*)d_in[32];
    const float* r2_m = (const float*)d_in[33], *r2_v  = (const float*)d_in[34];
    const float* r2_w = (const float*)d_in[35];
    const float* r3_b = (const float*)d_in[36];
    const float* r3_g = (const float*)d_in[37], *r3_be = (const float*)d_in[38];
    const float* r3_m = (const float*)d_in[39], *r3_v  = (const float*)d_in[40];
    const float* r3_w = (const float*)d_in[41];
    const float* r4_b = (const float*)d_in[42];
    const float* r4_w = (const float*)d_in[43];

    const int NB = in_sizes[3];            // batch (8)
    const int Np = in_sizes[1] / 3;        // points (200000)
    const size_t VOX3  = 64ull * 64 * 64;  // 262144 dense
    const size_t P66   = 66ull * 66 * 66;  // 287496 padded 64^3
    const size_t P34   = 34ull * 34 * 34;  // padded 32^3
    const size_t P18   = 18ull * 18 * 18;  // padded 16^3

    float* out_pre     = (float*)d_out;
    float* out_pre_sig = out_pre + (size_t)NB * VOX3;
    float* out_ref     = out_pre + 2ull * NB * VOX3;
    float* out_ref_sig = out_pre + 3ull * NB * VOX3;

    // ---- workspace carve-out ----
    char* ws = (char*)d_ws;
    size_t off = 0;
    auto alloc = [&](size_t bytes) -> char* {
        size_t o = (off + 255) & ~(size_t)255;
        off = o + bytes;
        return ws + o;
    };
    const size_t nA = (size_t)NB * 64 * P66;   // r1 out
    const size_t nB = (size_t)NB * 32 * P66;   // g2 / r2 out
    const size_t nCbig = (size_t)NB * 16 * P66;            // g3 / r3 out
    const size_t nCg1  = (size_t)NB * 64 * P34;            // g1 out
    const size_t nC = nCbig > nCg1 ? nCbig : nCg1;
    _Float16* actA  = (_Float16*)alloc(nA * 2);
    _Float16* actB  = (_Float16*)alloc(nB * 2);
    _Float16* actC  = (_Float16*)alloc(nC * 2);
    _Float16* in16  = (_Float16*)alloc((size_t)NB * P18 * 2);
    _Float16* updh  = (_Float16*)alloc((size_t)NB * P66 * 2);
    int*      winner= (int*)alloc((size_t)NB * VOX3 * 4);
    _Float16* w_g1  = (_Float16*)alloc(8ull * 64 * 32 * 2);
    _Float16* w_g2  = (_Float16*)alloc(8ull * 32 * 512 * 2);
    _Float16* w_g3  = (_Float16*)alloc(16ull * 864 * 2);
    _Float16* w_g4  = (_Float16*)alloc(16ull * 32 * 2);
    _Float16* w_r1  = (_Float16*)alloc(64ull * 32 * 2);
    _Float16* w_r2  = (_Float16*)alloc(32ull * 1728 * 2);
    _Float16* w_r3  = (_Float16*)alloc(16ull * 864 * 2);
    _Float16* w_r4  = (_Float16*)alloc(16ull * 32 * 2);
    int* t_g1 = (int*)alloc(32 * 4);
    int* t_g2 = (int*)alloc(512 * 4);
    int* t_g3 = (int*)alloc(864 * 4);
    int* t_g4 = (int*)alloc(32 * 4);
    int* t_r1 = (int*)alloc(32 * 4);
    int* t_r2 = (int*)alloc(1728 * 4);
    int* t_r3 = (int*)alloc(864 * 4);
    int* t_r4 = (int*)alloc(32 * 4);
    float* sc_g1 = (float*)alloc(64 * 4); float* sh_g1 = (float*)alloc(64 * 4);
    float* sc_g2 = (float*)alloc(32 * 4); float* sh_g2 = (float*)alloc(32 * 4);
    float* sc_g3 = (float*)alloc(16 * 4); float* sh_g3 = (float*)alloc(16 * 4);
    float* sc_g4 = (float*)alloc(16 * 4); float* sh_g4 = (float*)alloc(16 * 4);
    float* sc_r1 = (float*)alloc(64 * 4); float* sh_r1 = (float*)alloc(64 * 4);
    float* sc_r2 = (float*)alloc(32 * 4); float* sh_r2 = (float*)alloc(32 * 4);
    float* sc_r3 = (float*)alloc(16 * 4); float* sh_r3 = (float*)alloc(16 * 4);
    float* sc_r4 = (float*)alloc(16 * 4); float* sh_r4 = (float*)alloc(16 * 4);

    // ---- prep: weights, tables, BN folds ----
    prep_wconvt2<<<(8 * 64 * 32 + 255) / 256, 256, 0, stream>>>(g1_w, w_g1, 64, 1, 32, 64);
    prep_wconvt2<<<(8 * 32 * 512 + 255) / 256, 256, 0, stream>>>(g2_w, w_g2, 32, 64, 512, 32);
    prep_wconv<<<(16 * 864 + 255) / 256, 256, 0, stream>>>(g3_w, w_g3, 16, 32, 3, 864, 16, 1);
    prep_wconv<<<(16 * 32 + 255) / 256, 256, 0, stream>>>(g4_w, w_g4, 1, 16, 1, 32, 16, 1);
    prep_wconv<<<(64 * 32 + 255) / 256, 256, 0, stream>>>(r1_w, w_r1, 64, 1, 3, 32, 64, 0);
    prep_wconv<<<(32 * 1728 + 255) / 256, 256, 0, stream>>>(r2_w, w_r2, 32, 64, 3, 1728, 32, 0);
    prep_wconv<<<(16 * 864 + 255) / 256, 256, 0, stream>>>(r3_w, w_r3, 16, 32, 3, 864, 16, 0);
    prep_wconv<<<(16 * 32 + 255) / 256, 256, 0, stream>>>(r4_w, w_r4, 1, 16, 1, 32, 16, 0);
    prep_tab<<<1, 32, 0, stream>>>(t_g1, 8, 32, 2, 18);
    prep_tab<<<2, 256, 0, stream>>>(t_g2, 512, 512, 2, 34);
    prep_tab<<<4, 256, 0, stream>>>(t_g3, 864, 864, 3, 66);
    prep_tab<<<1, 32, 0, stream>>>(t_g4, 16, 32, 1, 66);
    prep_tab<<<1, 32, 0, stream>>>(t_r1, 27, 32, 3, 66);
    prep_tab<<<7, 256, 0, stream>>>(t_r2, 1728, 1728, 3, 66);
    prep_tab<<<4, 256, 0, stream>>>(t_r3, 864, 864, 3, 66);
    prep_tab<<<1, 32, 0, stream>>>(t_r4, 16, 32, 1, 66);
    prep_bn<<<1, 64, 0, stream>>>(g1_g, g1_be, g1_m, g1_v, g1_b, sc_g1, sh_g1, 64, 64);
    prep_bn<<<1, 64, 0, stream>>>(g2_g, g2_be, g2_m, g2_v, g2_b, sc_g2, sh_g2, 32, 32);
    prep_bn<<<1, 64, 0, stream>>>(g3_g, g3_be, g3_m, g3_v, g3_b, sc_g3, sh_g3, 16, 16);
    prep_bias<<<1, 64, 0, stream>>>(g4_b, sc_g4, sh_g4, 1, 16);
    prep_bn<<<1, 64, 0, stream>>>(r1_g, r1_be, r1_m, r1_v, r1_b, sc_r1, sh_r1, 64, 64);
    prep_bn<<<1, 64, 0, stream>>>(r2_g, r2_be, r2_m, r2_v, r2_b, sc_r2, sh_r2, 32, 32);
    prep_bn<<<1, 64, 0, stream>>>(r3_g, r3_be, r3_m, r3_v, r3_b, sc_r3, sh_r3, 16, 16);
    prep_bias<<<1, 64, 0, stream>>>(r4_b, sc_r4, sh_r4, 1, 16);

    // ---- input: f32 16^3 -> f16 padded 18^3 ----
    fill_h(in16, (size_t)NB * P18, stream);
    cvt_in_pad<<<(NB * 4096 + 255) / 256, 256, 0, stream>>>(vox_patch, in16, NB);

    ConvParams cp;

    // ---- g1: ConvT(1->64,k4,s2,p1), 16^3 -> 32^3, 8 parity calls ----
    fill_h(actC, nCg1, stream);
    for (int p = 0; p < 8; ++p) {
        const int pz = (p >> 2) & 1, py = (p >> 1) & 1, px = p & 1;
        cp = {};
        cp.in = in16; cp.w = w_g1 + (size_t)p * 64 * 32; cp.tab = t_g1;
        cp.scale = sc_g1; cp.shift = sh_g1; cp.outh = actC;
        cp.n_batch = NB; cp.Cin = 1; cp.Cout = 64; cp.Mtiles = 4;
        cp.Ws = 16; cp.Hs = 16; cp.Ds = 16; cp.Dp = 18; cp.Kpad = 32;
        cp.padz = 1 - pz; cp.pady = 1 - py; cp.padx = 1 - px;
        cp.sm = 2; cp.oz = pz; cp.oy = py; cp.ox = px;
        cp.Do_ = 32; cp.Dpo = 34; cp.lrelu = 1;
        launch_conv(cp, stream);
    }
    // ---- g2: ConvT(64->32,k4,s2,p1), 32^3 -> 64^3, 8 parity calls ----
    fill_h(actB, nB, stream);
    for (int p = 0; p < 8; ++p) {
        const int pz = (p >> 2) & 1, py = (p >> 1) & 1, px = p & 1;
        cp = {};
        cp.in = actC; cp.w = w_g2 + (size_t)p * 32 * 512; cp.tab = t_g2;
        cp.scale = sc_g2; cp.shift = sh_g2; cp.outh = actB;
        cp.n_batch = NB; cp.Cin = 64; cp.Cout = 32; cp.Mtiles = 2;
        cp.Ws = 32; cp.Hs = 32; cp.Ds = 32; cp.Dp = 34; cp.Kpad = 512;
        cp.padz = 1 - pz; cp.pady = 1 - py; cp.padx = 1 - px;
        cp.sm = 2; cp.oz = pz; cp.oy = py; cp.ox = px;
        cp.Do_ = 64; cp.Dpo = 66; cp.lrelu = 1;
        launch_conv(cp, stream);
    }
    // ---- g3: ConvT(32->16,k3,s1,p1) == flipped conv, 64^3 ----
    fill_h(actC, nCbig, stream);
    cp = {};
    cp.in = actB; cp.w = w_g3; cp.tab = t_g3;
    cp.scale = sc_g3; cp.shift = sh_g3; cp.outh = actC;
    cp.n_batch = NB; cp.Cin = 32; cp.Cout = 16; cp.Mtiles = 1;
    cp.Ws = 64; cp.Hs = 64; cp.Ds = 64; cp.Dp = 66; cp.Kpad = 864;
    cp.padz = 1; cp.pady = 1; cp.padx = 1;
    cp.sm = 1; cp.Do_ = 64; cp.Dpo = 66; cp.lrelu = 1;
    launch_conv(cp, stream);
    // ---- g4: 1x1x1 16->1 : vox_pre f32, sigmoid f32, sigmoid f16 (update) --
    fill_h(updh, (size_t)NB * P66, stream);
    cp = {};
    cp.in = actC; cp.w = w_g4; cp.tab = t_g4;
    cp.scale = sc_g4; cp.shift = sh_g4;
    cp.outf = out_pre; cp.outf_sig = out_pre_sig; cp.outh_sig = updh;
    cp.n_batch = NB; cp.Cin = 16; cp.Cout = 1; cp.Mtiles = 1;
    cp.Ws = 64; cp.Hs = 64; cp.Ds = 64; cp.Dp = 66; cp.Kpad = 32;
    cp.padz = 0; cp.pady = 0; cp.padx = 0;
    cp.sm = 1; cp.Do_ = 64; cp.Dpo = 66; cp.lrelu = 0;
    launch_conv(cp, stream);

    // ---- deterministic point scatter update ----
    {
        const int nw = NB * (int)VOX3;
        fill_i32<<<(nw + 255) / 256, 256, 0, stream>>>(winner, -1, nw);
        const int npts = NB * Np;
        scatter_pass1<<<(npts + 255) / 256, 256, 0, stream>>>(pc, patch_idx, winner, Np, NB);
        scatter_pass2<<<(npts + 255) / 256, 256, 0, stream>>>(pc, patch_idx, winner,
                                                             out_pre_sig, updh, Np, NB);
    }

    // ---- r1: Conv(1->64,k3,p1) ----
    fill_h(actA, nA, stream);
    cp = {};
    cp.in = updh; cp.w = w_r1; cp.tab = t_r1;
    cp.scale = sc_r1; cp.shift = sh_r1; cp.outh = actA;
    cp.n_batch = NB; cp.Cin = 1; cp.Cout = 64; cp.Mtiles = 4;
    cp.Ws = 64; cp.Hs = 64; cp.Ds = 64; cp.Dp = 66; cp.Kpad = 32;
    cp.padz = 1; cp.pady = 1; cp.padx = 1;
    cp.sm = 1; cp.Do_ = 64; cp.Dpo = 66; cp.lrelu = 1;
    launch_conv(cp, stream);
    // ---- r2: Conv(64->32,k3,p1) ----
    fill_h(actB, nB, stream);
    cp.in = actA; cp.w = w_r2; cp.tab = t_r2;
    cp.scale = sc_r2; cp.shift = sh_r2; cp.outh = actB;
    cp.Cin = 64; cp.Cout = 32; cp.Mtiles = 2; cp.Kpad = 1728;
    launch_conv(cp, stream);
    // ---- r3: Conv(32->16,k3,p1) ----
    fill_h(actC, nCbig, stream);
    cp.in = actB; cp.w = w_r3; cp.tab = t_r3;
    cp.scale = sc_r3; cp.shift = sh_r3; cp.outh = actC;
    cp.Cin = 32; cp.Cout = 16; cp.Mtiles = 1; cp.Kpad = 864;
    launch_conv(cp, stream);
    // ---- r4: 1x1x1 16->1 : vox_ref f32 + sigmoid f32 ----
    cp = {};
    cp.in = actC; cp.w = w_r4; cp.tab = t_r4;
    cp.scale = sc_r4; cp.shift = sh_r4;
    cp.outf = out_ref; cp.outf_sig = out_ref_sig;
    cp.n_batch = NB; cp.Cin = 16; cp.Cout = 1; cp.Mtiles = 1;
    cp.Ws = 64; cp.Hs = 64; cp.Ds = 64; cp.Dp = 66; cp.Kpad = 32;
    cp.padz = 0; cp.pady = 0; cp.padx = 0;
    cp.sm = 1; cp.Do_ = 64; cp.Dpo = 66; cp.lrelu = 0;
    launch_conv(cp, stream);
}
